// SelfAttentionV2_6983616824220
// MI455X (gfx1250) — matrix operand order repaired and checked by reading, compiled, not run on hardware
//
#include <hip/hip_runtime.h>

// Self-attention forward for MI455X (gfx1250, wave32, WMMA).
//   q = x@Wq.T ; k = x@Wk.T ; v = x@Wv.T      (f16 WMMA, f32 accum)
//   out = softmax(q@k.T / 32) @ v              (flash-style streaming softmax)
// Workspace layout (needs 48MB): Q[8192x1024] f16, K[8192x1024] f16, Vt[1024x8192] f16.

#define SEQL   8192
#define DMODEL 1024

typedef __attribute__((ext_vector_type(16))) _Float16 v16h;
typedef __attribute__((ext_vector_type(8)))  _Float16 v8h;
typedef __attribute__((ext_vector_type(8)))  float    v8f;
typedef __attribute__((ext_vector_type(4)))  float    v4f;

// log2(e) / sqrt(1024): softmax computed in base-2 (v_exp_f32 is exp2)
#define SCORE_SCALE 0.04508422f

static __device__ inline v16h cat16(v8h lo, v8h hi) {
  v16h r;
#pragma unroll
  for (int i = 0; i < 8; ++i) { r[i] = lo[i]; r[i + 8] = hi[i]; }
  return r;
}

// convert 8 contiguous f32 -> 8 f16
static __device__ inline v8h cvt8(const float* p) {
  v4f a = *(const v4f*)p;
  v4f b = *(const v4f*)(p + 4);
  v8h r;
#pragma unroll
  for (int i = 0; i < 4; ++i) { r[i] = (_Float16)a[i]; r[i + 4] = (_Float16)b[i]; }
  return r;
}

// A-matrix 16x32 f16 fragment (ISA 7.12.2): lane holds row m=lane%16,
// K pattern = {k0+8*hi .. +7} U {k0+16+8*hi .. +7}
static __device__ inline v16h loadA_f32(const float* row, int k0, int hi) {
  return cat16(cvt8(row + k0 + 8 * hi), cvt8(row + k0 + 16 + 8 * hi));
}
static __device__ inline v16h loadA_f16(const _Float16* row, int k0, int hi) {
  return cat16(*(const v8h*)(row + k0 + 8 * hi),
               *(const v8h*)(row + k0 + 16 + 8 * hi));
}
// B-matrix 32x16 f16 fragment: lane holds col n=lane%16,
// K pattern = {k0+8*hi .. +7} U {k0+16+8*hi .. +7} (same as A)
static __device__ inline v16h loadB_f32(const float* row, int k0, int hi) {
  return cat16(cvt8(row + k0 + 8 * hi), cvt8(row + k0 + 16 + 8 * hi));
}
static __device__ inline v16h loadB_f16(const _Float16* row, int k0, int hi) {
  return cat16(*(const v8h*)(row + k0 + 8 * hi),
               *(const v8h*)(row + k0 + 16 + 8 * hi));
}

static __device__ inline v8f wmma16(v16h a, v16h b, v8f c) {
  return __builtin_amdgcn_wmma_f32_16x16x32_f16(false, a, false, b,
                                                (short)0, c, false, false);
}

// ---------------------------------------------------------------------------
// Kernel 1: fused QKV projection.  grid = (512, 4, 3), block = 128 (4 waves).
// Wave computes a 16x64 output tile; z selects Q / K / V.
// Q,K stored row-major [s][d] f16; V stored transposed [d][s] f16.
// ---------------------------------------------------------------------------
__global__ __launch_bounds__(128) void qkv_proj_kernel(
    const float* __restrict__ x,  const float* __restrict__ Wq,
    const float* __restrict__ Wk, const float* __restrict__ Wv,
    _Float16* __restrict__ Qh, _Float16* __restrict__ Kh,
    _Float16* __restrict__ Vt) {
  const int lane = threadIdx.x & 31;
  const int wave = threadIdx.x >> 5;
  const int hi   = lane >> 4;     // which half of the wave
  const int lr   = lane & 15;
  const int m0   = blockIdx.x * 16;
  const int n0   = blockIdx.y * 256 + wave * 64;
  const int which = blockIdx.z;
  const float* W = (which == 0) ? Wq : (which == 1) ? Wk : Wv;

  const float* xrow = x + (size_t)(m0 + lr) * DMODEL;

  v8f acc[4] = {};
  for (int k0 = 0; k0 < DMODEL; k0 += 32) {
    v16h a = loadA_f32(xrow, k0, hi);
#pragma unroll
    for (int nf = 0; nf < 4; ++nf) {
      const float* wrow = W + (size_t)(n0 + nf * 16 + lr) * DMODEL;
      v16h b = loadB_f32(wrow, k0, hi);
      acc[nf] = wmma16(a, b, acc[nf]);   // reuse A across 4 B tiles
    }
  }

  // D layout: lane(n=lr, half hi), vgpr r -> row m = m0 + r + 8*hi
  if (which < 2) {
    _Float16* O = (which == 0) ? Qh : Kh;
#pragma unroll
    for (int nf = 0; nf < 4; ++nf) {
      const int n = n0 + nf * 16 + lr;
#pragma unroll
      for (int r = 0; r < 8; ++r)
        O[(size_t)(m0 + 8 * hi + r) * DMODEL + n] = (_Float16)acc[nf][r];
    }
  } else {
    // transposed store: 8 contiguous columns m -> one b128 per lane
#pragma unroll
    for (int nf = 0; nf < 4; ++nf) {
      const int n = n0 + nf * 16 + lr;
      v8h pk;
#pragma unroll
      for (int r = 0; r < 8; ++r) pk[r] = (_Float16)acc[nf][r];
      *(v8h*)(Vt + (size_t)n * SEQL + m0 + 8 * hi) = pk;
    }
  }
}

// ---------------------------------------------------------------------------
// Kernel 2: flash attention.  grid = 512 (one block per 16 query rows),
// block = 256 (8 waves).  Per 128-key iteration:
//   - wave w computes the 16x16 score tile for its 16 keys (Q from LDS)
//   - ALL 256 threads update the streaming softmax (16 threads/row, 8 cols
//     each, __shfl_xor reductions inside 16-lane groups), emit P (f16)
//   - every wave accumulates P @ Vt for its private 128-wide d-slice
// ---------------------------------------------------------------------------
__global__ __launch_bounds__(256) void flash_attn_kernel(
    const _Float16* __restrict__ Qh, const _Float16* __restrict__ Kh,
    const _Float16* __restrict__ Vt, float* __restrict__ out) {
  __shared__ __align__(16) _Float16 Qs[16 * DMODEL];   // 32 KB
  __shared__ __align__(16) float    Ss[16 * 128];      //  8 KB
  __shared__ __align__(16) _Float16 Ps[16 * 128];      //  4 KB
  __shared__ float rowm[16], rowl[16], rowa[16];

  const int tid  = threadIdx.x;
  const int lane = tid & 31;
  const int wave = tid >> 5;
  const int hi   = lane >> 4;
  const int lr   = lane & 15;
  const int m0   = blockIdx.x * 16;
  const int dw   = wave * 128;          // this wave's d-slice

  // stage the 16x1024 f16 Q tile into LDS (b128 copies)
  {
    const v8h* src = (const v8h*)(Qh + (size_t)m0 * DMODEL);
    v8h* dst = (v8h*)Qs;
    for (int i = tid; i < 16 * DMODEL / 8; i += 256) dst[i] = src[i];
  }
  if (tid < 16) { rowm[tid] = -1e30f; rowl[tid] = 0.0f; }
  __syncthreads();

  v8f o[8] = {};

  for (int j0 = 0; j0 < SEQL; j0 += 128) {
    const int jw = j0 + wave * 16;      // this wave's 16 keys

    if (j0 + 128 < SEQL)                // warm L2/L0 for next key tile
      __builtin_prefetch(Kh + (size_t)(jw + 128 + lr) * DMODEL, 0, 1);

    // ---- scores: S = Q(16x1024) @ K_tile^T, f32 accum --------------------
    v8f s = {};
    const _Float16* qrow = Qs + (size_t)lr * DMODEL;
    const _Float16* krow = Kh + (size_t)(jw + lr) * DMODEL;
    for (int k0 = 0; k0 < DMODEL; k0 += 32) {
      v16h a = loadA_f16(qrow, k0, hi);   // ds_load_b128 x2
      v16h b = loadB_f16(krow, k0, hi);   // global_load_b128 x2 (L2 hit)
      s = wmma16(a, b, s);
    }
#pragma unroll
    for (int r = 0; r < 8; ++r)
      Ss[(r + 8 * hi) * 128 + wave * 16 + lr] = s[r] * SCORE_SCALE;
    __syncthreads();

    // ---- parallel streaming softmax: 16 threads/row, 8 cols/thread ------
    {
      const int r = tid >> 4;          // query row 0..15 (2 rows per wave)
      const int g = tid & 15;          // column group within the row
      const float* srow = Ss + r * 128 + g * 8;
      v4f s0 = *(const v4f*)srow;
      v4f s1 = *(const v4f*)(srow + 4);
      float lm = s0[0];
#pragma unroll
      for (int i = 1; i < 4; ++i) lm = fmaxf(lm, s0[i]);
#pragma unroll
      for (int i = 0; i < 4; ++i) lm = fmaxf(lm, s1[i]);
#pragma unroll
      for (int off = 1; off < 16; off <<= 1)      // row-max across 16 lanes
        lm = fmaxf(lm, __shfl_xor(lm, off, 32));
      const float mold = rowm[r];      // read-before-write: wave lockstep
      const float mnew = fmaxf(mold, lm);
      v8h p;
      float ls = 0.0f;
#pragma unroll
      for (int i = 0; i < 4; ++i) {
        float p0 = exp2f(s0[i] - mnew);            // base-2 softmax
        float p1 = exp2f(s1[i] - mnew);
        p[i] = (_Float16)p0; p[i + 4] = (_Float16)p1;
        ls += p0 + p1;
      }
      *(v8h*)(Ps + r * 128 + g * 8) = p;           // ds_store_b128
#pragma unroll
      for (int off = 1; off < 16; off <<= 1)       // row-sum across 16 lanes
        ls += __shfl_xor(ls, off, 32);
      if (g == 0) {
        const float alpha = exp2f(mold - mnew);    // 0 on first iteration
        rowl[r] = rowl[r] * alpha + ls;
        rowm[r] = mnew;
        rowa[r] = alpha;
      }
    }
    __syncthreads();

    // ---- rescale accumulators, then O += P(16x128) @ Vt-slice ------------
    float ar[8];
#pragma unroll
    for (int r = 0; r < 8; ++r) ar[r] = rowa[r + 8 * hi];
#pragma unroll
    for (int nf = 0; nf < 8; ++nf)
#pragma unroll
      for (int r = 0; r < 8; ++r) o[nf][r] *= ar[r];

    const _Float16* prow = Ps + (size_t)lr * 128;
#pragma unroll
    for (int kt = 0; kt < 4; ++kt) {
      v16h a = loadA_f16(prow, kt * 32, hi);            // P fragment from LDS
#pragma unroll
      for (int nf = 0; nf < 8; ++nf) {
        const _Float16* vrow = Vt + (size_t)(dw + nf * 16 + lr) * SEQL + j0;
        v16h b = loadB_f16(vrow, kt * 32, hi);          // Vt fragment (L2 hit)
        o[nf] = wmma16(a, b, o[nf]);
      }
    }
    // Ps/rowa of this iteration are only rewritten after the next iteration's
    // first barrier, so two barriers per iteration are sufficient.
  }

  // ---- epilogue: divide by running sum, store f32 ------------------------
  float lr8[8];
#pragma unroll
  for (int r = 0; r < 8; ++r) lr8[r] = 1.0f / rowl[r + 8 * hi];
#pragma unroll
  for (int nf = 0; nf < 8; ++nf) {
    const int n = dw + nf * 16 + lr;
#pragma unroll
    for (int r = 0; r < 8; ++r)
      out[(size_t)(m0 + r + 8 * hi) * DMODEL + n] = o[nf][r] * lr8[r];
  }
}

extern "C" void kernel_launch(void* const* d_in, const int* in_sizes, int n_in,
                              void* d_out, int out_size, void* d_ws, size_t ws_size,
                              hipStream_t stream) {
  const float* x  = (const float*)d_in[0];
  const float* wq = (const float*)d_in[1];
  const float* wk = (const float*)d_in[2];
  const float* wv = (const float*)d_in[3];

  _Float16* Qh = (_Float16*)d_ws;                       // 16 MB
  _Float16* Kh = Qh + (size_t)SEQL * DMODEL;            // 16 MB
  _Float16* Vt = Kh + (size_t)SEQL * DMODEL;            // 16 MB (transposed V)
  float* out = (float*)d_out;

  dim3 gproj(SEQL / 16, DMODEL / 256, 3);
  qkv_proj_kernel<<<gproj, 128, 0, stream>>>(x, wq, wk, wv, Qh, Kh, Vt);
  flash_attn_kernel<<<SEQL / 16, 256, 0, stream>>>(Qh, Kh, Vt, out);
}
